// CountMambaModel_24283745091714
// MI455X (gfx1250) — matrix operand order repaired and checked by reading, not compiled, not run on hardware
//
#include <hip/hip_runtime.h>
#include <math.h>

// ---------------------------------------------------------------------------
// CountMamba forward for MI455X (gfx1250), wave32 / WMMA.
// Dense contractions -> v_wmma_f32_16x16x32_bf16 (bf16 in, f32 acc), with a
// vectorized interior fast path (global_load_b128 -> cvt_pk_bf16 -> ds_store).
// Sequential selective scan keeps full (P=64 x N=128) state in registers.
// ---------------------------------------------------------------------------

typedef __attribute__((ext_vector_type(16))) __bf16 v16bf;
typedef __attribute__((ext_vector_type(8)))  float  v8f;

#define TM 64
#define TN 64
#define TK 32

// Generic C = A(MxK) @ B(KxN) with optional im2col B and fused epilogues.
// bmode: 0 = dense row-major B ; 1 = im2col over (C,Lin) input, kk = c*5+k, shift l+k-4
// epi:   0 none | 1 rowbias+BN+ReLU (p0=convb,p1=bn_g,p2=bn_b) | 2 rowbias (p0)
//        3 colbias (p0) | 4 accumulate into C (residual)
__global__ __launch_bounds__(256)
void gemm_wmma(const float* __restrict__ A, const float* __restrict__ Bm,
               float* __restrict__ C,
               int M, int N, int K, int lda, int ldb, int ldc,
               long long sA, long long sB, long long sC,
               int bmode, int Lin,
               int epi, const float* __restrict__ p0,
               const float* __restrict__ p1, const float* __restrict__ p2) {
  __shared__ __attribute__((aligned(16))) __bf16 As[TM][TK];
  __shared__ __attribute__((aligned(16))) __bf16 Bs[TN][TK];   // transposed [n][k]

  const int tid  = threadIdx.x;
  const int wave = tid >> 5;
  const int lane = tid & 31;
  const int mt    = wave & 3;        // 4 M-tiles of 16
  const int nhalf = wave >> 2;       // 2 N-halves of 32 (2x16 tiles each)
  const int m0 = blockIdx.x * TM;
  const int n0 = blockIdx.y * TN;
  const int z  = blockIdx.z;

  const float* Ab = A  + (long long)z * sA;
  const float* Bb = Bm + (long long)z * sB;
  float*       Cb = C  + (long long)z * sC;

  // uniform fast-path predicate (whole block in-bounds, full K chunks)
  const bool fastA = (m0 + TM <= M) && ((K & (TK - 1)) == 0);
  const bool fastB = (n0 + TN <= N) && ((K & (TK - 1)) == 0) &&
                     (bmode == 0 || n0 >= 4);

  v8f acc0 = {0.f,0.f,0.f,0.f,0.f,0.f,0.f,0.f};
  v8f acc1 = {0.f,0.f,0.f,0.f,0.f,0.f,0.f,0.f};

  for (int kc = 0; kc < K; kc += TK) {
    // ---- stage A tile (64x32) as bf16 ----
    if (fastA) {
#pragma unroll
      for (int i = 0; i < 2; ++i) {             // 512 float4 / 256 threads
        int idx = tid + i * 256;
        int r = idx >> 3, c4 = (idx & 7) << 2;
        const float4 v = *(const float4*)(Ab + (long long)(m0 + r) * lda + kc + c4);
        As[r][c4 + 0] = (__bf16)v.x;
        As[r][c4 + 1] = (__bf16)v.y;
        As[r][c4 + 2] = (__bf16)v.z;
        As[r][c4 + 3] = (__bf16)v.w;
      }
    } else {
#pragma unroll
      for (int i = 0; i < 8; ++i) {
        int t = tid + i * 256;
        int r = t >> 5, c = t & 31;
        int gm = m0 + r, gk = kc + c;
        float v = (gm < M && gk < K) ? Ab[(long long)gm * lda + gk] : 0.f;
        As[r][c] = (__bf16)v;
      }
    }
    // ---- stage B tile (32k x 64n), transposed to [n][k] ----
    if (fastB) {
#pragma unroll
      for (int i = 0; i < 2; ++i) {             // 512 float4 / 256 threads
        int idx = tid + i * 256;
        int kk = idx >> 4, n4 = (idx & 15) << 2;
        int gk = kc + kk;
        float w0, w1, w2, w3;
        if (bmode == 0) {
          const float4 v = *(const float4*)(Bb + (long long)gk * ldb + n0 + n4);
          w0 = v.x; w1 = v.y; w2 = v.z; w3 = v.w;
        } else {
          int c5 = gk / 5, kof = gk - c5 * 5;
          const float* src = Bb + (long long)c5 * Lin + (n0 + n4 + kof - 4);
          w0 = src[0]; w1 = src[1]; w2 = src[2]; w3 = src[3];
        }
        Bs[n4 + 0][kk] = (__bf16)w0;
        Bs[n4 + 1][kk] = (__bf16)w1;
        Bs[n4 + 2][kk] = (__bf16)w2;
        Bs[n4 + 3][kk] = (__bf16)w3;
      }
    } else {
#pragma unroll
      for (int i = 0; i < 8; ++i) {
        int t = tid + i * 256;
        int kk = t >> 6, nn = t & 63;
        int gk = kc + kk, gn = n0 + nn;
        float v = 0.f;
        if (gk < K && gn < N) {
          if (bmode == 0) {
            v = Bb[(long long)gk * ldb + gn];
          } else {                       // im2col: causal conv, taps k=0..4, pad 4
            int c5 = gk / 5, kof = gk - c5 * 5;
            int l  = gn + kof - 4;
            if (l >= 0 && l < Lin) v = Bb[(long long)c5 * Lin + l];
          }
        }
        Bs[nn][kk] = (__bf16)v;
      }
    }
    __syncthreads();

    // ---- fragments: contiguous k order, lane half g holds k = 16g..16g+15 (A and B alike) ----
    const int g  = lane >> 4;
    const int ml  = mt * 16 + (lane & 15);
    const int nl0 = nhalf * 32 + (lane & 15);
    union U { v16bf v; uint4 q[2]; } ua, ub0, ub1;
    const uint4* ap = (const uint4*)(&As[ml][0]);
    ua.q[0] = ap[2 * g];      // K = g*16 .. g*16+7
    ua.q[1] = ap[2 * g + 1];  // K = g*16+8 .. g*16+15
    const uint4* bp0 = (const uint4*)(&Bs[nl0][0]);
    ub0.q[0] = bp0[2 * g];    // K = g*16 .. g*16+7
    ub0.q[1] = bp0[2 * g + 1];
    const uint4* bp1 = (const uint4*)(&Bs[nl0 + 16][0]);
    ub1.q[0] = bp1[2 * g];
    ub1.q[1] = bp1[2 * g + 1];

    acc0 = __builtin_amdgcn_wmma_f32_16x16x32_bf16(
        false, ua.v, false, ub0.v, (short)0, acc0, false, false);
    acc1 = __builtin_amdgcn_wmma_f32_16x16x32_bf16(
        false, ua.v, false, ub1.v, (short)0, acc1, false, false);
    __syncthreads();
  }

  // ---- epilogue: D layout = VGPR r -> row r (+8 for lanes 16-31), col = lane%16
  const int g = lane >> 4;
  const float bnr = rsqrtf(1.0f + 1e-5f);
#pragma unroll
  for (int t = 0; t < 2; ++t) {
    const v8f accv = t ? acc1 : acc0;
    const int col = n0 + nhalf * 32 + t * 16 + (lane & 15);
#pragma unroll
    for (int r = 0; r < 8; ++r) {
      int row = m0 + mt * 16 + g * 8 + r;
      if (row < M && col < N) {
        float v = accv[r];
        long long o = (long long)row * ldc + col;
        if (epi == 1) {
          float sc = p1[row] * bnr;
          v = (v + p0[row]) * sc + p2[row];
          v = v > 0.f ? v : 0.f;
        } else if (epi == 2) v += p0[row];
        else if (epi == 3)   v += p0[col];
        else if (epi == 4)   v += Cb[o];
        Cb[o] = v;
      }
    }
  }
}

// ---------------------------------------------------------------------------
__global__ void maxpool_k(const float* __restrict__ in, float* __restrict__ out,
                          long long BC, int Lout, int k) {
  long long i = (long long)blockIdx.x * 256 + threadIdx.x;
  if (i >= BC * Lout) return;
  long long bc = i / Lout; int lo = (int)(i % Lout);
  const float* p = in + bc * (long long)Lout * k + (long long)lo * k;
  float m = p[0];
  for (int j = 1; j < k; ++j) m = fmaxf(m, p[j]);
  out[i] = m;
}

// u[b,l,e] = (l==0) ? cls+pos0 : h[b,e,l-1] + pos[l]
__global__ void assemble_u(const float* __restrict__ hp, const float* __restrict__ cls,
                           const float* __restrict__ pos, float* __restrict__ u,
                           int Lseq, int E, int Lh, long long total) {
  long long i = (long long)blockIdx.x * 256 + threadIdx.x;
  if (i >= total) return;
  int e = (int)(i % E);
  long long t = i / E;
  int l = (int)(t % Lseq);
  long long b = t / Lseq;
  float v;
  if (l == 0) v = cls[e] + pos[e];
  else        v = hp[(b * E + e) * (long long)Lh + (l - 1)] + pos[(long long)l * E + e];
  u[i] = v;
}

// depthwise causal conv (K=4) + SiLU on xBC slice of zxbcdt
__global__ void xbc_conv_silu(const float* __restrict__ zx, const float* __restrict__ cwl,
                              const float* __restrict__ cbl, float* __restrict__ xbc,
                              long long rows, int L, int ldin, int coff, int CD) {
  long long i = (long long)blockIdx.x * 256 + threadIdx.x;
  if (i >= rows * CD) return;
  int c = (int)(i % CD);
  long long row = i / CD;
  int l = (int)(row % L);
  float s = cbl[c];
#pragma unroll
  for (int k = 0; k < 4; ++k) {
    int dl = l - 3 + k;
    if (dl >= 0) s += cwl[c * 4 + k] * zx[(row - 3 + k) * (long long)ldin + coff + c];
  }
  float sig = 1.f / (1.f + expf(-s));
  xbc[i] = s * sig;
}

// dt = softplus(raw + bias), dA = exp(-exp(A_log)*dt)
__global__ void dt_kernel(const float* __restrict__ zx, const float* __restrict__ dtb,
                          const float* __restrict__ alog, float* __restrict__ dt,
                          float* __restrict__ dA, long long rows, int ldin, int doff, int H) {
  long long i = (long long)blockIdx.x * 256 + threadIdx.x;
  if (i >= rows * H) return;
  int h = (int)(i % H);
  long long row = i / H;
  float raw = zx[row * (long long)ldin + doff + h] + dtb[h];
  float sp = raw > 20.f ? raw : log1pf(expf(raw));
  dt[i] = sp;
  dA[i] = expf(-expf(alog[h]) * sp);
}

// selective scan: one block per (b,h); state[64][128] lives in registers
// thread t: p = t/4, n-range = (t%4)*32 .. +31  (32 state floats/thread)
__global__ __launch_bounds__(256)
void scan_kernel(const float* __restrict__ xbc, const float* __restrict__ dtv,
                 const float* __restrict__ dAv, const float* __restrict__ Dh,
                 float* __restrict__ ys, int L, int H) {
  const int b = blockIdx.x / H, h = blockIdx.x % H;
  const int tid = threadIdx.x;
  const int p = tid >> 2, ng = tid & 3;
  const int nb = ng * 32;
  const int CD = 768;
  float hst[32];
#pragma unroll
  for (int i = 0; i < 32; ++i) hst[i] = 0.f;
  __shared__ float xs_s[64], B_s[128], C_s[128];
  const float dcoef = Dh[h];

  for (int l = 0; l < L; ++l) {
    long long row = (long long)b * L + l;
    const float* xr = xbc + row * CD;
    for (int i = tid; i < 320; i += 256) {
      if (i < 64)       xs_s[i]       = xr[h * 64 + i];
      else if (i < 192) B_s[i - 64]   = xr[512 + (i - 64)];
      else              C_s[i - 192]  = xr[640 + (i - 192)];
    }
    __syncthreads();
    float da  = dAv[row * H + h];
    float dtx = dtv[row * H + h] * xs_s[p];
    float part = 0.f;
#pragma unroll
    for (int i = 0; i < 32; ++i) {
      float hv = hst[i] * da + dtx * B_s[nb + i];
      hst[i] = hv;
      part += hv * C_s[nb + i];
    }
    part += __shfl_xor(part, 1, 32);
    part += __shfl_xor(part, 2, 32);
    if (ng == 0) ys[row * 512 + h * 64 + p] = part + dcoef * xs_s[p];
    __syncthreads();
  }
}

// y = (ys * silu(z)) * rsqrt(mean(y^2)+eps) * norm_w  (in place on ys)
__global__ __launch_bounds__(256)
void gate_rms(const float* __restrict__ zx, const float* __restrict__ nw,
              float* __restrict__ y, int ldin, int D) {
  long long row = blockIdx.x;
  int tid = threadIdx.x;
  float z0 = zx[row * (long long)ldin + tid];
  float z1 = zx[row * (long long)ldin + tid + 256];
  float v0 = y[row * (long long)D + tid]       * (z0 / (1.f + expf(-z0)));
  float v1 = y[row * (long long)D + tid + 256] * (z1 / (1.f + expf(-z1)));
  float ss = v0 * v0 + v1 * v1;
#pragma unroll
  for (int o = 16; o > 0; o >>= 1) ss += __shfl_xor(ss, o, 32);
  __shared__ float red[8];
  if ((tid & 31) == 0) red[tid >> 5] = ss;
  __syncthreads();
  float tot = 0.f;
#pragma unroll
  for (int i = 0; i < 8; ++i) tot += red[i];
  float scale = rsqrtf(tot / (float)D + 1e-5f);
  y[row * (long long)D + tid]       = v0 * scale * nw[tid];
  y[row * (long long)D + tid + 256] = v1 * scale * nw[tid + 256];
}

__global__ __launch_bounds__(256)
void layernorm_k(const float* __restrict__ u, const float* __restrict__ g,
                 const float* __restrict__ bta, float* __restrict__ out) {
  long long row = blockIdx.x;
  int tid = threadIdx.x;
  const int E = 256;
  float x = u[row * E + tid];
  float s = x;
#pragma unroll
  for (int o = 16; o > 0; o >>= 1) s += __shfl_xor(s, o, 32);
  __shared__ float red[8];
  if ((tid & 31) == 0) red[tid >> 5] = s;
  __syncthreads();
  float tot = 0.f;
#pragma unroll
  for (int i = 0; i < 8; ++i) tot += red[i];
  float mean = tot / 256.f;
  float d = x - mean;
  float vs = d * d;
#pragma unroll
  for (int o = 16; o > 0; o >>= 1) vs += __shfl_xor(vs, o, 32);
  __syncthreads();
  if ((tid & 31) == 0) red[tid >> 5] = vs;
  __syncthreads();
  float vtot = 0.f;
#pragma unroll
  for (int i = 0; i < 8; ++i) vtot += red[i];
  out[row * E + tid] = d * rsqrtf(vtot / 256.f + 1e-5f) * g[tid] + bta[tid];
}

__global__ void pool_k(const float* __restrict__ hn, const int* __restrict__ idx,
                       float* __restrict__ pooled, int Lseq, int E) {
  int b = blockIdx.x, e = threadIdx.x;
  int agg = idx[b] / 6;
  float s = 0.f;
  for (int l = 0; l <= agg; ++l)
    s += hn[((long long)b * Lseq + 1 + l) * E + e];
  pooled[b * E + e] = s / (float)(agg + 1);
}

// ---------------------------------------------------------------------------
extern "C" void kernel_launch(void* const* d_in, const int* in_sizes, int n_in,
                              void* d_out, int out_size, void* d_ws, size_t ws_size,
                              hipStream_t stream) {
  const float* x         = (const float*)d_in[0];
  const int*   idx       = (const int*)  d_in[1];
  const float* patch_w   = (const float*)d_in[2];
  const float* patch_b   = (const float*)d_in[3];
  const float* conv_w    = (const float*)d_in[4];
  const float* conv_b    = (const float*)d_in[5];
  const float* bn_g      = (const float*)d_in[6];
  const float* bn_b      = (const float*)d_in[7];
  const float* cls_token = (const float*)d_in[8];
  const float* pos_embed = (const float*)d_in[9];
  const float* Wi        = (const float*)d_in[10];
  const float* cw        = (const float*)d_in[11];
  const float* cb        = (const float*)d_in[12];
  const float* dt_bias   = (const float*)d_in[13];
  const float* A_log     = (const float*)d_in[14];
  const float* Dh        = (const float*)d_in[15];
  const float* norm_w    = (const float*)d_in[16];
  const float* Wo        = (const float*)d_in[17];
  const float* ln_g      = (const float*)d_in[18];
  const float* ln_b      = (const float*)d_in[19];
  const float* fc_w      = (const float*)d_in[20];
  const float* fc_b      = (const float*)d_in[21];

  const int Bb = 16, E = 256, L0 = 12288;
  const int depth = 8, H = 8, d_inner = 512, CD = 768, dproj = 1288;
  const int Lseq = L0 / 6 + 1;                 // 2049
  const long long BL = (long long)Bb * Lseq;   // 32784

  auto cdiv = [](long long a, long long b) { return (unsigned)((a + b - 1) / b); };

  // workspace layout (floats) ~506 MB
  float* ws     = (float*)d_ws;
  float* bufA   = ws;
  float* bufB   = bufA + (long long)Bb * E * L0;
  float* u      = bufB + (long long)Bb * E * L0;
  float* dtb    = u    + BL * E;
  float* dAb    = dtb  + BL * H;
  float* ys     = dAb  + BL * H;
  float* pooled = ys   + BL * d_inner;
  float* zx     = bufA;   // reused after conv stage
  float* xbc    = bufB;   // reused after conv stage
  float* lnout  = bufB;   // reused after mamba stage

  // 1) patch embed: per-batch GEMM  patch_w(256x8) @ x_b(8xL0) + patch_b
  {
    dim3 g(cdiv(E, TM), cdiv(L0, TN), Bb);
    gemm_wmma<<<g, 256, 0, stream>>>(patch_w, x, bufA, E, L0, 8, 8, L0, L0,
        0, (long long)8 * L0, (long long)E * L0, 0, L0, 2, patch_b, nullptr, nullptr);
  }
  // 2) conv stack: conv as GEMM(weights(Ex1280), im2col(1280xL)) + BN + ReLU
  auto conv = [&](const float* in, float* out, int Lc, int ly) {
    dim3 g(cdiv(E, TM), cdiv(Lc, TN), Bb);
    gemm_wmma<<<g, 256, 0, stream>>>(conv_w + (long long)ly * E * E * 5, in, out,
        E, Lc, E * 5, E * 5, 0, Lc,
        0, (long long)E * Lc, (long long)E * Lc, 1, Lc,
        1, conv_b + ly * E, bn_g + ly * E, bn_b + ly * E);
  };
  conv(bufA, bufB, L0, 0);
  conv(bufB, bufA, L0, 1);
  int L1 = L0 / 3;                                     // 4096
  maxpool_k<<<cdiv((long long)Bb * E * L1, 256), 256, 0, stream>>>(bufA, bufB, (long long)Bb * E, L1, 3);
  conv(bufB, bufA, L1, 2);
  conv(bufA, bufB, L1, 3);
  int L2 = L1 / 2;                                     // 2048
  maxpool_k<<<cdiv((long long)Bb * E * L2, 256), 256, 0, stream>>>(bufB, bufA, (long long)Bb * E, L2, 2);
  // 3) assemble token sequence (cls + pos embed, transpose to (b,l,e))
  assemble_u<<<cdiv(BL * E, 256), 256, 0, stream>>>(bufA, cls_token, pos_embed, u, Lseq, E, L2, BL * E);

  // 4) Mamba2 layers
  for (int ly = 0; ly < depth; ++ly) {
    {   // zxbcdt = u @ Wi
      dim3 g(cdiv(BL, TM), cdiv(dproj, TN), 1);
      gemm_wmma<<<g, 256, 0, stream>>>(u, Wi + (long long)ly * E * dproj, zx,
          (int)BL, dproj, E, E, dproj, dproj, 0, 0, 0, 0, 0, 0, nullptr, nullptr, nullptr);
    }
    xbc_conv_silu<<<cdiv(BL * CD, 256), 256, 0, stream>>>(
        zx, cw + (long long)ly * CD * 4, cb + ly * CD, xbc, BL, Lseq, dproj, d_inner, CD);
    dt_kernel<<<cdiv(BL * H, 256), 256, 0, stream>>>(
        zx, dt_bias + ly * H, A_log + ly * H, dtb, dAb, BL, dproj, d_inner + CD, H);
    scan_kernel<<<Bb * H, 256, 0, stream>>>(xbc, dtb, dAb, Dh + ly * H, ys, Lseq, H);
    gate_rms<<<(unsigned)BL, 256, 0, stream>>>(zx, norm_w + ly * d_inner, ys, dproj, d_inner);
    {   // u += ys @ Wo  (residual)
      dim3 g(cdiv(BL, TM), cdiv(E, TN), 1);
      gemm_wmma<<<g, 256, 0, stream>>>(ys, Wo + (long long)ly * d_inner * E, u,
          (int)BL, E, d_inner, d_inner, E, E, 0, 0, 0, 0, 0, 4, nullptr, nullptr, nullptr);
    }
  }

  // 5) head
  layernorm_k<<<(unsigned)BL, 256, 0, stream>>>(u, ln_g, ln_b, lnout);
  pool_k<<<Bb, 256, 0, stream>>>(lnout, idx, pooled, Lseq, E);
  {
    dim3 g(cdiv(Bb, TM), cdiv(100, TN), 1);
    gemm_wmma<<<g, 256, 0, stream>>>(pooled, fc_w, (float*)d_out,
        Bb, 100, E, E, 100, 100, 0, 0, 0, 0, 0, 3, fc_b, nullptr, nullptr);
  }
}
